// SpatioTemporalGNN_56092272886105
// MI455X (gfx1250) — matrix operand order repaired and checked by reading, not compiled, not run on hardware
//
#include <hip/hip_runtime.h>
#include <hip/hip_bf16.h>
#include <math.h>
#include <stdint.h>

#define N_NODES 50000            // == 3125 * 16, so no row-tile remainder
#define N_EDGES 800000
#define T_STEPS 6
#define F_IN    64
#define HID     48
#define NCLS    2
#define LN_EPS  1e-5f

typedef __attribute__((ext_vector_type(16))) _Float16 v16h;
typedef __attribute__((ext_vector_type(8)))  float    v8f;

// ---------------------------------------------------------------- utilities
__global__ void fill_kernel(float* __restrict__ p, float v, int n) {
  int i = blockIdx.x * blockDim.x + threadIdx.x;
  if (i < n) p[i] = v;
}

// deg[col[e]] += w[e]  (deg pre-filled with 1.0 for the self loop)
__global__ void deg_kernel(const long long* __restrict__ col,
                           const float* __restrict__ w,
                           float* __restrict__ deg, int e) {
  int i = blockIdx.x * blockDim.x + threadIdx.x;
  if (i < e) atomicAdd(&deg[(int)col[i]], w[i]);
}

// in-place: dis = deg>0 ? rsqrt(max(deg,1e-12)) : 0
__global__ void dis_kernel(float* __restrict__ d, int n) {
  int i = blockIdx.x * blockDim.x + threadIdx.x;
  if (i < n) {
    float dg = d[i];
    d[i] = dg > 0.0f ? rsqrtf(fmaxf(dg, 1e-12f)) : 0.0f;
  }
}

__global__ void norm_kernel(const long long* __restrict__ row,
                            const long long* __restrict__ col,
                            const float* __restrict__ w,
                            const float* __restrict__ dis,
                            float* __restrict__ norm, int e) {
  int i = blockIdx.x * blockDim.x + threadIdx.x;
  if (i < e) norm[i] = dis[(int)row[i]] * w[i] * dis[(int)col[i]];
}

__global__ void normself_kernel(const float* __restrict__ dis,
                                float* __restrict__ ns, int n) {
  int i = blockIdx.x * blockDim.x + threadIdx.x;
  if (i < n) { float d = dis[i]; ns[i] = d * d; }
}

// -------------------------------------------------- weight pre-pack (once)
// Pack B(k,n) = Bsrc[k*bSK + n*bSN] into the WMMA 32x16 f16 B-fragment layout:
//   out[((tileN*nChunks + chunk)*32 + lane)*16 + h] = B(chunk*32 + h + (h & 8) + laneHi*8,
//                                                       tileN*16 + laneLo)
// Zero-padded past Kdim. One thread per (tileN, chunk, lane).
__global__ void pack_b_kernel(const float* __restrict__ B, int bSK, int bSN,
                              int Kdim, int nCols, int nChunks,
                              _Float16* __restrict__ out) {
  int idx = blockIdx.x * blockDim.x + threadIdx.x;
  int total = (nCols / 16) * nChunks * 32;
  if (idx >= total) return;
  int lane   = idx & 31;
  int chunk  = (idx >> 5) % nChunks;
  int tn     = idx / (32 * nChunks);
  int laneLo = lane & 15;
  int laneHi = lane >> 4;
  int col    = tn * 16 + laneLo;
  _Float16* o = out + (size_t)idx * 16;
  for (int h = 0; h < 16; ++h) {
    int kk = chunk * 32 + h + (h & 8) + laneHi * 8;
    o[h] = (kk < Kdim) ? (_Float16)B[(size_t)kk * bSK + (size_t)col * bSN]
                       : (_Float16)0.f;
  }
}

// ---------------------------------------------------- WMMA dense projection
// C[N_NODES x NCOLS] = A[N_NODES x KDIM] * B (pre-packed f16 fragments).
// One wave per 16x16 output tile. KDIM in {48,64}: K-chunks fully unrolled,
// all bounds resolved at compile time (no EXEC predication in the hot loop).
template <int KDIM, int NCOLS>
__global__ void gemm_wmma_kernel(const float* __restrict__ A,
                                 const _Float16* __restrict__ Bp,
                                 float* __restrict__ C, int nTiles) {
  constexpr int NCHUNK  = (KDIM + 31) / 32;
  constexpr int NTILESN = NCOLS / 16;
  int tile = (int)((blockIdx.x * blockDim.x + threadIdx.x) >> 5);
  if (tile >= nTiles) return;                 // wave-uniform exit
  int tm = tile / NTILESN, tn = tile % NTILESN;
  int lane   = threadIdx.x & 31;
  int laneLo = lane & 15;
  int laneHi = lane >> 4;
  const float* __restrict__ arow = A + (size_t)(tm * 16 + laneLo) * KDIM;

  v8f acc = {0.f, 0.f, 0.f, 0.f, 0.f, 0.f, 0.f, 0.f};
#pragma unroll
  for (int c = 0; c < NCHUNK; ++c) {
    const int k0 = c * 32;
    v16h av;
    // A fragment: halves 0..7  <- K = k0 + 8*laneHi + 0..7   (contiguous)
    //             halves 8..15 <- K = k0 + 16 + 8*laneHi + 0..7
    const float4 p0 = *(const float4*)(arow + k0 + laneHi * 8);
    const float4 p1 = *(const float4*)(arow + k0 + laneHi * 8 + 4);
    av[0] = (_Float16)p0.x; av[1] = (_Float16)p0.y;
    av[2] = (_Float16)p0.z; av[3] = (_Float16)p0.w;
    av[4] = (_Float16)p1.x; av[5] = (_Float16)p1.y;
    av[6] = (_Float16)p1.z; av[7] = (_Float16)p1.w;
    if (k0 + 32 <= KDIM) {                    // compile-time after unroll
      const float4 p2 = *(const float4*)(arow + k0 + 16 + laneHi * 8);
      const float4 p3 = *(const float4*)(arow + k0 + 16 + laneHi * 8 + 4);
      av[8]  = (_Float16)p2.x; av[9]  = (_Float16)p2.y;
      av[10] = (_Float16)p2.z; av[11] = (_Float16)p2.w;
      av[12] = (_Float16)p3.x; av[13] = (_Float16)p3.y;
      av[14] = (_Float16)p3.z; av[15] = (_Float16)p3.w;
    } else {
#pragma unroll
      for (int h = 8; h < 16; ++h) av[h] = (_Float16)0.f;
    }
    // B fragment: 16 halves, pre-packed contiguously per lane (32B aligned)
    const v16h bv =
        *(const v16h*)(Bp + ((size_t)(tn * NCHUNK + c) * 32 + lane) * 16);
    acc = __builtin_amdgcn_wmma_f32_16x16x32_f16(
        /*neg_a=*/false, av, /*neg_b=*/false, bv,
        /*c_mod=*/(short)0, acc, /*reuse_a=*/false, /*reuse_b=*/false);
  }
  // C/D layout: vgpr v -> row tm*16 + v + 8*laneHi, lane -> column
#pragma unroll
  for (int v = 0; v < 8; ++v)
    C[(size_t)(tm * 16 + v + laneHi * 8) * NCOLS + tn * 16 + laneLo] = acc[v];
}

// ------------------------------------------------------- GCN message passing
// agg[i,f] = b[f] + xw[i,f] * normself[i]        (self loop + bias)
__global__ void self_bias_kernel(const float* __restrict__ xw,
                                 const float* __restrict__ ns,
                                 const float* __restrict__ b,
                                 float* __restrict__ agg, int total) {
  int idx = blockIdx.x * blockDim.x + threadIdx.x;
  if (idx >= total) return;
  int node = idx / HID, f = idx - node * HID;
  agg[idx] = b[f] + xw[idx] * ns[node];
}

// agg[col,f] += xw[row,f] * norm[e]
__global__ void scatter_kernel(const long long* __restrict__ row,
                               const long long* __restrict__ col,
                               const float* __restrict__ norm,
                               const float* __restrict__ xw,
                               float* __restrict__ agg, int e) {
  long long idx = (long long)blockIdx.x * blockDim.x + threadIdx.x;
  if (idx >= (long long)e * HID) return;
  int ei = (int)(idx / HID), f = (int)(idx - (long long)ei * HID);
  int r = (int)row[ei], c = (int)col[ei];
  atomicAdd(&agg[(size_t)c * HID + f], xw[(size_t)r * HID + f] * norm[ei]);
}

// out = relu(LN(x)*g+be) [+ residual]; one wave32 per node, 48 features
__global__ void ln_relu_kernel(const float* __restrict__ x,
                               const float* __restrict__ g,
                               const float* __restrict__ be,
                               const float* __restrict__ residual,
                               float* __restrict__ out, int n) {
  int wave = (int)((blockIdx.x * blockDim.x + threadIdx.x) >> 5);
  int lane = threadIdx.x & 31;
  if (wave >= n) return;
  const size_t base = (size_t)wave * HID;
  float a = x[base + lane];
  bool hi = lane < (HID - 32);                 // lanes 0..15 also own f=lane+32
  float b = hi ? x[base + lane + 32] : 0.0f;
  float s = a + b;
#pragma unroll
  for (int off = 16; off; off >>= 1) s += __shfl_xor(s, off, 32);
  float mu = s * (1.0f / HID);
  float da = a - mu, db = b - mu;
  float sq = da * da + (hi ? db * db : 0.0f);
#pragma unroll
  for (int off = 16; off; off >>= 1) sq += __shfl_xor(sq, off, 32);
  float rinv = rsqrtf(sq * (1.0f / HID) + LN_EPS);
  float y0 = fmaxf(da * rinv * g[lane] + be[lane], 0.0f);
  if (residual) y0 += residual[base + lane];
  out[base + lane] = y0;
  if (hi) {
    float y1 = fmaxf(db * rinv * g[lane + 32] + be[lane + 32], 0.0f);
    if (residual) y1 += residual[base + lane + 32];
    out[base + lane + 32] = y1;
  }
}

// ------------------------------------------------------------ GRU gate math
__global__ void gru_gate_kernel(const float* __restrict__ gi,
                                const float* __restrict__ gh,
                                const float* __restrict__ bih,
                                const float* __restrict__ bhh,
                                const float* __restrict__ hprev,
                                float* __restrict__ hnew, int n) {
  int idx = blockIdx.x * blockDim.x + threadIdx.x;
  if (idx >= n * HID) return;
  int node = idx / HID, f = idx - node * HID;
  const float* gir = gi + (size_t)node * (3 * HID);
  const float* ghr = gh + (size_t)node * (3 * HID);
  float ir = gir[f]           + bih[f];
  float iz = gir[HID + f]     + bih[HID + f];
  float in = gir[2 * HID + f] + bih[2 * HID + f];
  float hr = ghr[f]           + bhh[f];
  float hz = ghr[HID + f]     + bhh[HID + f];
  float hn = ghr[2 * HID + f] + bhh[2 * HID + f];
  float r = 1.0f / (1.0f + __expf(-(ir + hr)));
  float z = 1.0f / (1.0f + __expf(-(iz + hz)));
  float nv = tanhf(in + r * hn);
  float hp = hprev[idx];
  hnew[idx] = (1.0f - z) * nv + z * hp;
}

// ------------------------------------------------------- classifier (tiny)
__global__ void classifier_kernel(const float* __restrict__ zf,
                                  const float* __restrict__ Wc1,
                                  const float* __restrict__ bc1,
                                  const float* __restrict__ Wc2,
                                  const float* __restrict__ bc2,
                                  float* __restrict__ out, int n) {
  int i = blockIdx.x * blockDim.x + threadIdx.x;
  if (i >= n) return;
  const float* zi = zf + (size_t)i * HID;
  float hid[HID / 2];
#pragma unroll
  for (int j = 0; j < HID / 2; ++j) {
    float s = bc1[j];
    for (int k = 0; k < HID; ++k) s += zi[k] * Wc1[k * (HID / 2) + j];
    hid[j] = fmaxf(s, 0.0f);
  }
#pragma unroll
  for (int c = 0; c < NCLS; ++c) {
    float s = bc2[c];
    for (int j = 0; j < HID / 2; ++j) s += hid[j] * Wc2[j * NCLS + c];
    out[(size_t)i * NCLS + c] = s;
  }
}

// ----------------------------------------------------------------- launcher
extern "C" void kernel_launch(void* const* d_in, const int* in_sizes, int n_in,
                              void* d_out, int out_size, void* d_ws, size_t ws_size,
                              hipStream_t stream) {
  (void)in_sizes; (void)n_in; (void)out_size; (void)ws_size;
  const float*     x_seq = (const float*)d_in[0];
  const long long* eidx  = (const long long*)d_in[1];
  const float*     ew    = (const float*)d_in[2];
  const float* W0  = (const float*)d_in[3];
  const float* b0  = (const float*)d_in[4];
  const float* g0  = (const float*)d_in[5];
  const float* be0 = (const float*)d_in[6];
  const float* W1  = (const float*)d_in[7];
  const float* b1  = (const float*)d_in[8];
  const float* g1  = (const float*)d_in[9];
  const float* be1 = (const float*)d_in[10];
  const float* Wih = (const float*)d_in[11];
  const float* Whh = (const float*)d_in[12];
  const float* bih = (const float*)d_in[13];
  const float* bhh = (const float*)d_in[14];
  const float* Wc1 = (const float*)d_in[15];
  const float* bc1 = (const float*)d_in[16];
  const float* Wc2 = (const float*)d_in[17];
  const float* bc2 = (const float*)d_in[18];

  const long long* row = eidx;            // edge_index[0]
  const long long* col = eidx + N_EDGES;  // edge_index[1]

  // ---- workspace layout (floats, then 64B-aligned f16 packed weights) ----
  float* ws  = (float*)d_ws;
  float* dis = ws;                               // N   (deg -> dis in place)
  float* nrm = dis + N_NODES;                    // E
  float* nsl = nrm + N_EDGES;                    // N
  float* xw  = nsl + N_NODES;                    // N*HID
  float* agg = xw  + (size_t)N_NODES * HID;      // N*HID
  float* h1  = agg + (size_t)N_NODES * HID;      // N*HID
  float* h2  = h1  + (size_t)N_NODES * HID;      // N*HID
  float* gi  = h2  + (size_t)N_NODES * HID;      // N*3*HID
  float* gh  = gi  + (size_t)N_NODES * 3 * HID;  // N*3*HID
  float* hA  = gh  + (size_t)N_NODES * 3 * HID;  // N*HID
  float* hB  = hA  + (size_t)N_NODES * HID;      // N*HID
  uintptr_t pk = ((uintptr_t)(hB + (size_t)N_NODES * HID) + 63u) & ~(uintptr_t)63u;
  _Float16* bp0 = (_Float16*)pk;                 // W0  : 3 tiles * 2 chunks * 512
  _Float16* bp1 = bp0 + 3 * 2 * 32 * 16;         // W1  : 3 * 2 * 512
  _Float16* bpI = bp1 + 3 * 2 * 32 * 16;         // Wih^T: 9 * 2 * 512
  _Float16* bpH = bpI + 9 * 2 * 32 * 16;         // Whh^T: 9 * 2 * 512

  const int BLK = 256;
  const int gN  = (N_NODES + BLK - 1) / BLK;
  const int gE  = (N_EDGES + BLK - 1) / BLK;
  const int gNH = (N_NODES * HID + BLK - 1) / BLK;
  const long long eh = (long long)N_EDGES * HID;
  const int gEH = (int)((eh + BLK - 1) / BLK);
  const int gLN = (N_NODES + (BLK / 32) - 1) / (BLK / 32);

  // --- pack weights into WMMA f16 B-fragments (tiny, once per launch) ---
  pack_b_kernel<<<1, 256, 0, stream>>>(W0, HID, 1, F_IN, HID, 2, bp0);  // 192 thr
  pack_b_kernel<<<1, 256, 0, stream>>>(W1, HID, 1, HID, HID, 2, bp1);   // 192 thr
  pack_b_kernel<<<3, 256, 0, stream>>>(Wih, 1, HID, HID, 3 * HID, 2, bpI); // 576
  pack_b_kernel<<<3, 256, 0, stream>>>(Whh, 1, HID, HID, 3 * HID, 2, bpH); // 576

  // --- GCN normalization (self-loop weight folded into deg init = 1.0) ---
  fill_kernel<<<gN, BLK, 0, stream>>>(dis, 1.0f, N_NODES);
  deg_kernel<<<gE, BLK, 0, stream>>>(col, ew, dis, N_EDGES);
  dis_kernel<<<gN, BLK, 0, stream>>>(dis, N_NODES);
  norm_kernel<<<gE, BLK, 0, stream>>>(row, col, ew, dis, nrm, N_EDGES);
  normself_kernel<<<gN, BLK, 0, stream>>>(dis, nsl, N_NODES);

  // --- GRU state ---
  fill_kernel<<<gNH, BLK, 0, stream>>>(hA, 0.0f, N_NODES * HID);
  float* hprev = hA;
  float* hnext = hB;

  const int rowT = N_NODES / 16;                 // 3125 exact
  const int t48  = rowT * (HID / 16);            // 9375 tiles
  const int t144 = rowT * (3 * HID / 16);        // 28125 tiles
  const int g48  = (t48  + (BLK / 32) - 1) / (BLK / 32);
  const int g144 = (t144 + (BLK / 32) - 1) / (BLK / 32);

  for (int t = 0; t < T_STEPS; ++t) {
    const float* xt = x_seq + (size_t)t * N_NODES * F_IN;

    // GCN layer 0: h1 = relu(LN(scatter(xt@W0) + b0))
    gemm_wmma_kernel<F_IN, HID><<<g48, BLK, 0, stream>>>(xt, bp0, xw, t48);
    self_bias_kernel<<<gNH, BLK, 0, stream>>>(xw, nsl, b0, agg, N_NODES * HID);
    scatter_kernel<<<gEH, BLK, 0, stream>>>(row, col, nrm, xw, agg, N_EDGES);
    ln_relu_kernel<<<gLN, BLK, 0, stream>>>(agg, g0, be0, nullptr, h1, N_NODES);

    // GCN layer 1: h2 = relu(LN(scatter(h1@W1) + b1)) + h1
    gemm_wmma_kernel<HID, HID><<<g48, BLK, 0, stream>>>(h1, bp1, xw, t48);
    self_bias_kernel<<<gNH, BLK, 0, stream>>>(xw, nsl, b1, agg, N_NODES * HID);
    scatter_kernel<<<gEH, BLK, 0, stream>>>(row, col, nrm, xw, agg, N_EDGES);
    ln_relu_kernel<<<gLN, BLK, 0, stream>>>(agg, g1, be1, h1, h2, N_NODES);

    // GRU projections (W^T absorbed into the packed-B layout)
    gemm_wmma_kernel<HID, 3 * HID><<<g144, BLK, 0, stream>>>(h2, bpI, gi, t144);
    gemm_wmma_kernel<HID, 3 * HID><<<g144, BLK, 0, stream>>>(hprev, bpH, gh, t144);
    gru_gate_kernel<<<gNH, BLK, 0, stream>>>(gi, gh, bih, bhh, hprev, hnext,
                                             N_NODES);
    float* tmp = hprev; hprev = hnext; hnext = tmp;
  }

  classifier_kernel<<<gN, BLK, 0, stream>>>(hprev, Wc1, bc1, Wc2, bc2,
                                            (float*)d_out, N_NODES);
}
